// PointNetPPEnc_67010079752487
// MI455X (gfx1250) — matrix operand order repaired and checked by reading, not compiled, not run on hardware
//
#include <hip/hip_runtime.h>
#include <stdint.h>

// ---------------------------------------------------------------------------
// PointNet++ MSG encoder for MI455X (gfx1250, wave32, WMMA 16x16x32 f16)
// Register-blocked 2x2 WMMA tiles per wave: each A/B fragment feeds 2 WMMAs,
// 4 independent accumulator chains per k-step.
// ---------------------------------------------------------------------------

typedef __attribute__((ext_vector_type(16))) _Float16 v16h;
typedef __attribute__((ext_vector_type(8)))  float    v8f;

#define BN_EPS 1e-5f
#define BATCH 8

static inline int a32h(int x) { return (x + 31) & ~31; }
static inline size_t a256(size_t x) { return (x + 255) & ~(size_t)255; }

// ---------------------------------------------------------------------------
// WMMA fragment loaders.
// A (16x32 f16): lane L holds row M = L&15; halves 0..7 = k0..k0+7,
// halves 8..15 = k0+16..k0+23, where k0 = (L>=16 ? 8 : 0) (+32 per k-tile).
// B (32x16 f16): lane L holds column N = L&15, same k order as A
// (weights packed transposed [N][K], loaded with load_a_frag).
// ---------------------------------------------------------------------------
__device__ inline v16h load_a_frag(const _Float16* p) {
  v16h r;
  const char* s = (const char*)__builtin_assume_aligned(p, 16);
  __builtin_memcpy(&r, s, 16);                                    // ds_load_b128
  __builtin_memcpy(reinterpret_cast<char*>(&r) + 16, s + 32, 16); // +16 halves
  return r;
}
__device__ inline v16h load_b_frag(const _Float16* p) {
  v16h r;
  __builtin_memcpy(&r, __builtin_assume_aligned(p, 16), 32);      // 2x b128
  return r;
}
__device__ inline v8f wmma_f16(v16h a, v16h b, v8f c) {
  return __builtin_amdgcn_wmma_f32_16x16x32_f16(false, a, false, b, (short)0, c,
                                                false, false);
}
__device__ inline void atomic_max_pos(float* p, float v) {
  // exact for v >= 0 against a zero-initialized buffer
  atomicMax(reinterpret_cast<int*>(p), __float_as_int(v));
}

// ---------------------------------------------------------------------------
// One MLP layer over a [Krows x cinpad] f16 activation tile in LDS.
// D = relu(scale * (A @ W) + bias). If out_gmax != nullptr, the row dimension
// is max-pooled via global atomic max (final layer of each SA stage);
// otherwise the f16 result is written to the LDS ping-pong buffer.
// 2x2 register blocking: mt_n and nt_n are always even here.
// ---------------------------------------------------------------------------
__device__ inline void run_layer(const _Float16* __restrict__ inb,
                                 _Float16* __restrict__ outb,
                                 const _Float16* __restrict__ w,
                                 const float* __restrict__ sb, // [scale|bias] x coutpad
                                 int Krows, int cinpad, int cout, int coutpad,
                                 float* out_gmax, int relu) {
  const int lane = threadIdx.x & 31;
  const int wave = threadIdx.x >> 5;
  const int nw   = blockDim.x >> 5;
  const int m_lo = lane & 15;
  const int hi   = (lane >> 4) & 1;
  const int mt2_n = Krows >> 5;   // pairs of 16-row tiles
  const int nt2_n = coutpad >> 5; // pairs of 16-col tiles
  const int kt_n  = cinpad >> 5;
  for (int t = wave; t < mt2_n * nt2_n; t += nw) {
    const int mt = (t / nt2_n) * 2;
    const int nt = (t - (t / nt2_n) * nt2_n) * 2;
    v8f acc00 = {}, acc01 = {}, acc10 = {}, acc11 = {};
    const _Float16* arow0 = inb + (size_t)(mt * 16 + m_lo) * cinpad + hi * 8;
    const _Float16* arow1 = arow0 + (size_t)16 * cinpad;
    const _Float16* wrow  = w + (size_t)(nt * 16 + m_lo) * cinpad + hi * 8;
    for (int kt = 0; kt < kt_n; ++kt) {
      v16h a0 = load_a_frag(arow0 + kt * 32);
      v16h a1 = load_a_frag(arow1 + kt * 32);
      v16h b0 = load_a_frag(wrow + kt * 32);
      v16h b1 = load_a_frag(wrow + (size_t)16 * cinpad + kt * 32);
      acc00 = wmma_f16(a0, b0, acc00);
      acc01 = wmma_f16(a0, b1, acc01);
      acc10 = wmma_f16(a1, b0, acc10);
      acc11 = wmma_f16(a1, b1, acc11);
    }
    const int col0 = nt * 16 + m_lo;
    const int col1 = col0 + 16;
    const float sc0 = sb[col0], bi0 = sb[coutpad + col0];
    const float sc1 = sb[col1], bi1 = sb[coutpad + col1];
#pragma unroll
    for (int r = 0; r < 8; ++r) {
      const int row0 = mt * 16 + r + hi * 8;
      const int row1 = row0 + 16;
      float v00 = fmaf(acc00[r], sc0, bi0);
      float v01 = fmaf(acc01[r], sc1, bi1);
      float v10 = fmaf(acc10[r], sc0, bi0);
      float v11 = fmaf(acc11[r], sc1, bi1);
      if (relu) {
        v00 = fmaxf(v00, 0.0f); v01 = fmaxf(v01, 0.0f);
        v10 = fmaxf(v10, 0.0f); v11 = fmaxf(v11, 0.0f);
      }
      if (out_gmax) {
        if (col0 < cout) {
          atomic_max_pos(&out_gmax[col0], v00);
          atomic_max_pos(&out_gmax[col0], v10);
        }
        if (col1 < cout) {
          atomic_max_pos(&out_gmax[col1], v01);
          atomic_max_pos(&out_gmax[col1], v11);
        }
      } else {
        outb[(size_t)row0 * coutpad + col0] = (_Float16)v00;
        outb[(size_t)row0 * coutpad + col1] = (_Float16)v01;
        outb[(size_t)row1 * coutpad + col0] = (_Float16)v10;
        outb[(size_t)row1 * coutpad + col1] = (_Float16)v11;
      }
    }
  }
}

// ---------------------------------------------------------------------------
// Weight / scale-bias preprocessing
// ---------------------------------------------------------------------------
__global__ void prep_weight(const float* __restrict__ W, int cin, int cout,
                            int kpad, int coutpad, _Float16* __restrict__ w16) {
  int total = kpad * coutpad;
  for (int e = blockIdx.x * blockDim.x + threadIdx.x; e < total;
       e += gridDim.x * blockDim.x) {
    int k = e / coutpad, c = e - k * coutpad;
    float v = (k < cin && c < cout) ? W[(size_t)k * cout + c] : 0.0f;
    w16[(size_t)c * kpad + k] = (_Float16)v;
  }
}

__global__ void prep_sb(const float* __restrict__ b, const float* __restrict__ g,
                        const float* __restrict__ be, const float* __restrict__ m,
                        const float* __restrict__ v, int cout, int coutpad,
                        float* __restrict__ sb) {
  int c = blockIdx.x * blockDim.x + threadIdx.x;
  if (c >= coutpad) return;
  float sc = 0.0f, bi = 0.0f;
  if (c < cout) {
    if (g) {
      float inv = rsqrtf(v[c] + BN_EPS);
      sc = g[c] * inv;
      bi = (b[c] - m[c]) * sc + be[c];
    } else {
      sc = 1.0f;
      bi = b[c];
    }
  }
  sb[c] = sc;
  sb[coutpad + c] = bi;
}

// ---------------------------------------------------------------------------
// Farthest point sampling: one block per batch, sequential over npoint steps.
// ---------------------------------------------------------------------------
__global__ void fps_kernel(const float* __restrict__ xyz, int Nsrc, int npoint,
                           float* __restrict__ dist, int* __restrict__ idx_out) {
  const int b = blockIdx.x;
  const int t = threadIdx.x;
  const float* p = xyz + (size_t)b * Nsrc * 3;
  float* d = dist + (size_t)b * Nsrc;
  __shared__ float s_val[256];
  __shared__ int   s_idx[256];
  __shared__ int   s_far;
  for (int j = t; j < Nsrc; j += 256) d[j] = 1e10f;
  if (t == 0) s_far = 0;
  __syncthreads();
  for (int it = 0; it < npoint; ++it) {
    const int far = s_far;
    if (t == 0) idx_out[b * npoint + it] = far;
    const float cx = p[far * 3 + 0], cy = p[far * 3 + 1], cz = p[far * 3 + 2];
    float best = -1.0f; int bi = 0;
    for (int j = t; j < Nsrc; j += 256) {
      float dx = p[j * 3 + 0] - cx;
      float dy = p[j * 3 + 1] - cy;
      float dz = p[j * 3 + 2] - cz;
      float dd = dx * dx + dy * dy + dz * dz;
      float nd = fminf(d[j], dd);
      d[j] = nd;
      if (nd > best) { best = nd; bi = j; }
    }
    s_val[t] = best; s_idx[t] = bi;
    __syncthreads();
    for (int s = 128; s > 0; s >>= 1) {
      if (t < s && s_val[t + s] > s_val[t]) {
        s_val[t] = s_val[t + s]; s_idx[t] = s_idx[t + s];
      }
      __syncthreads();
    }
    if (t == 0) s_far = s_idx[0];
    __syncthreads();
  }
}

__global__ void gather_xyz(const float* __restrict__ src, const int* __restrict__ idx,
                           int NP, int Nsrc, float* __restrict__ out) {
  int t = blockIdx.x * blockDim.x + threadIdx.x;
  if (t >= BATCH * NP) return;
  int b = t / NP;
  int n = idx[t];
  const float* s = src + ((size_t)b * Nsrc + n) * 3;
  out[t * 3 + 0] = s[0];
  out[t * 3 + 1] = s[1];
  out[t * 3 + 2] = s[2];
}

// ---------------------------------------------------------------------------
// Ball query: first K neighbors (index order) within radius; pad with first.
// ---------------------------------------------------------------------------
__global__ void ball_query_kernel(const float* __restrict__ xyz,
                                  const float* __restrict__ new_xyz,
                                  int Nsrc, int NP, int K, float r2,
                                  int* __restrict__ idx) {
  int t = blockIdx.x * blockDim.x + threadIdx.x;
  if (t >= BATCH * NP) return;
  int b = t / NP;
  const float* q = new_xyz + (size_t)t * 3;
  const float qx = q[0], qy = q[1], qz = q[2];
  const float* src = xyz + (size_t)b * Nsrc * 3;
  int* o = idx + (size_t)t * K;
  int cnt = 0, first = 0;
  for (int j = 0; j < Nsrc && cnt < K; ++j) {
    __builtin_prefetch(src + 3 * (j + 96), 0, 1); // global_prefetch_b8
    float dx = src[j * 3 + 0] - qx;
    float dy = src[j * 3 + 1] - qy;
    float dz = src[j * 3 + 2] - qz;
    if (dx * dx + dy * dy + dz * dz <= r2) {
      if (cnt == 0) first = j;
      o[cnt++] = j;
    }
  }
  for (; cnt < K; ++cnt) o[cnt] = first;
}

// ---------------------------------------------------------------------------
// Fused SA stage: gather -> 3x(WMMA matmul + BN + ReLU) -> max-pool (atomic).
// One block per (batch, center) group; activations live in LDS as f16.
// mode 0: feature = xyz[n]-center (sa1)
// mode 1: feature = [feat[n] (Cfeat), xyz[n]-center (3)] (sa2)
// mode 2: feature = [xyz[row] (3), feat[row] (Cfeat)], identity idx (sa3)
// ---------------------------------------------------------------------------
__global__ void sa_group_mlp(
    const float* __restrict__ xyz_src, const float* __restrict__ feat_src,
    const float* __restrict__ new_xyz, const int* __restrict__ idx,
    int NP, int Nsrc, int Krows, int Cfeat, int mode,
    const _Float16* __restrict__ w0, const float* __restrict__ sb0,
    int c0in_pad, int c0out, int c0out_pad,
    const _Float16* __restrict__ w1, const float* __restrict__ sb1,
    int c1out, int c1out_pad,
    const _Float16* __restrict__ w2, const float* __restrict__ sb2,
    int c2out, int c2out_pad,
    int ldsB_off_halfs,
    float* __restrict__ out, int out_stride, int out_coff) {
  extern __shared__ _Float16 smem[];
  _Float16* bufA = smem;
  _Float16* bufB = smem + ldsB_off_halfs;

  const int b = blockIdx.x / NP;
  const int c = blockIdx.x - b * NP;

  float cx = 0.f, cy = 0.f, cz = 0.f;
  if (mode < 2) {
    const float* q = new_xyz + (size_t)(b * NP + c) * 3;
    cx = q[0]; cy = q[1]; cz = q[2];
  }

  // Stage grouped input into LDS as f16, zero-padded to c0in_pad.
  const int total = Krows * c0in_pad;
  for (int e = threadIdx.x; e < total; e += blockDim.x) {
    const int row = e / c0in_pad;
    const int col = e - row * c0in_pad;
    float v = 0.0f;
    if (mode == 0) {
      if (col < 3) {
        int n = idx[(size_t)(b * NP + c) * Krows + row];
        const float* s = xyz_src + ((size_t)b * Nsrc + n) * 3;
        v = s[col] - (col == 0 ? cx : (col == 1 ? cy : cz));
      }
    } else if (mode == 1) {
      int n = idx[(size_t)(b * NP + c) * Krows + row];
      if (col < Cfeat) {
        v = feat_src[((size_t)b * Nsrc + n) * Cfeat + col];
      } else if (col < Cfeat + 3) {
        int k = col - Cfeat;
        const float* s = xyz_src + ((size_t)b * Nsrc + n) * 3;
        v = s[k] - (k == 0 ? cx : (k == 1 ? cy : cz));
      }
    } else {
      int n = row;
      if (col < 3) v = xyz_src[((size_t)b * Nsrc + n) * 3 + col];
      else if (col < 3 + Cfeat)
        v = feat_src[((size_t)b * Nsrc + n) * Cfeat + (col - 3)];
    }
    bufA[e] = (_Float16)v;
  }
  __syncthreads();

  run_layer(bufA, bufB, w0, sb0, Krows, c0in_pad, c0out, c0out_pad, nullptr, 1);
  __syncthreads();
  run_layer(bufB, bufA, w1, sb1, Krows, c0out_pad, c1out, c1out_pad, nullptr, 1);
  __syncthreads();
  float* og = out + (size_t)(b * NP + c) * out_stride + out_coff;
  run_layer(bufA, nullptr, w2, sb2, Krows, c1out_pad, c2out, c2out_pad, og, 1);
}

// ---------------------------------------------------------------------------
// Dense FC layer: one block per 16-row M tile (rows = 8 -> one block).
// 1x2 N-blocking: each A fragment feeds two WMMAs.
// ---------------------------------------------------------------------------
__global__ void dense_layer(const float* __restrict__ in, int rows, int cinreal,
                            int cinpad, const _Float16* __restrict__ w,
                            const float* __restrict__ sb, int cout, int coutpad,
                            float* __restrict__ out, int relu) {
  extern __shared__ _Float16 sA[];
  const int r0 = blockIdx.x * 16;
  for (int e = threadIdx.x; e < 16 * cinpad; e += blockDim.x) {
    const int row = e / cinpad;
    const int col = e - row * cinpad;
    float v = 0.0f;
    if (r0 + row < rows && col < cinreal)
      v = in[(size_t)(r0 + row) * cinreal + col];
    sA[e] = (_Float16)v;
  }
  __syncthreads();
  const int lane = threadIdx.x & 31;
  const int wave = threadIdx.x >> 5;
  const int nw   = blockDim.x >> 5;
  const int m_lo = lane & 15;
  const int hi   = (lane >> 4) & 1;
  const int nt2_n = coutpad >> 5;
  const int kt_n  = cinpad >> 5;
  for (int t = wave; t < nt2_n; t += nw) {
    const int nt = t * 2;
    v8f acc0 = {}, acc1 = {};
    const _Float16* arow = sA + (size_t)m_lo * cinpad + hi * 8;
    const _Float16* wrow = w + (size_t)(nt * 16 + m_lo) * cinpad + hi * 8;
    for (int kt = 0; kt < kt_n; ++kt) {
      v16h a  = load_a_frag(arow + kt * 32);
      v16h b0 = load_a_frag(wrow + kt * 32);
      v16h b1 = load_a_frag(wrow + (size_t)16 * cinpad + kt * 32);
      acc0 = wmma_f16(a, b0, acc0);
      acc1 = wmma_f16(a, b1, acc1);
    }
    const int col0 = nt * 16 + m_lo;
    const int col1 = col0 + 16;
    const float sc0 = sb[col0], bi0 = sb[coutpad + col0];
    const float sc1 = sb[col1], bi1 = sb[coutpad + col1];
#pragma unroll
    for (int r = 0; r < 8; ++r) {
      const int grow = r0 + r + hi * 8;
      float v0 = fmaf(acc0[r], sc0, bi0);
      float v1 = fmaf(acc1[r], sc1, bi1);
      if (relu) { v0 = fmaxf(v0, 0.0f); v1 = fmaxf(v1, 0.0f); }
      if (grow < rows) {
        if (col0 < cout) out[(size_t)grow * cout + col0] = v0;
        if (col1 < cout) out[(size_t)grow * cout + col1] = v1;
      }
    }
  }
}

// ---------------------------------------------------------------------------
// Host orchestration
// ---------------------------------------------------------------------------
struct LayerDesc {
  int Wi, bi, bei, gi, mi, vi;
  int cin, cout;
  size_t w_off, sb_off; // byte offsets in d_ws
};

extern "C" void kernel_launch(void* const* d_in, const int* in_sizes, int n_in,
                              void* d_out, int out_size, void* d_ws, size_t ws_size,
                              hipStream_t stream) {
  (void)in_sizes; (void)n_in; (void)out_size;
  char* ws = (char*)d_ws;

  // ---- input index map (jax tree order: sorted dict keys) ----
  // 0: xyz
  // 1-4: bn1 {be,g,m,v}   5-8: bn2 {be,g,m,v}
  // 9-10: fc1 {W,b}  11-12: fc2  13-14: fc3
  // 15-68: sa1 (3 mlps x 3 layers x {W,b,be,g,m,v})
  // 69-104: sa2 (2 x 3 x 6)   105-122: sa3 (3 x 6)
  const int sa1d[3][4] = {{3,32,32,64},{3,64,64,128},{3,64,96,128}};
  const int sa2d[2][4] = {{323,128,128,256},{323,128,196,256}};
  const int sa3d[4]    = {515,256,512,1024};

  LayerDesc L[21];
  int li = 0;
  for (int m = 0; m < 3; ++m)
    for (int l = 0; l < 3; ++l) {
      int base = 15 + m * 18 + l * 6;
      L[li++] = {base, base+1, base+2, base+3, base+4, base+5,
                 sa1d[m][l], sa1d[m][l+1], 0, 0};
    }
  for (int m = 0; m < 2; ++m)
    for (int l = 0; l < 3; ++l) {
      int base = 69 + m * 18 + l * 6;
      L[li++] = {base, base+1, base+2, base+3, base+4, base+5,
                 sa2d[m][l], sa2d[m][l+1], 0, 0};
    }
  for (int l = 0; l < 3; ++l) {
    int base = 105 + l * 6;
    L[li++] = {base, base+1, base+2, base+3, base+4, base+5,
               sa3d[l], sa3d[l+1], 0, 0};
  }
  L[li++] = {9, 10, 1, 2, 3, 4, 1024, 512, 0, 0};     // fc1 + bn1
  L[li++] = {11, 12, 5, 6, 7, 8, 512, 256, 0, 0};     // fc2 + bn2
  L[li++] = {13, 14, -1, -1, -1, -1, 256, 128, 0, 0}; // fc3 (no BN)

  // ---- workspace layout ----
  size_t cur = 0;
  auto take = [&](size_t bytes) { size_t o = cur; cur = a256(cur + bytes); return o; };
  const size_t off_dist  = take((size_t)BATCH * 8192 * 4);
  const size_t off_idxf1 = take((size_t)BATCH * 512 * 4);
  const size_t off_nx1   = take((size_t)BATCH * 512 * 3 * 4);
  const size_t off_bq    = take((size_t)BATCH * 512 * 128 * 4); // reused per scale
  const size_t off_l1    = take((size_t)BATCH * 512 * 320 * 4);
  const size_t off_idxf2 = take((size_t)BATCH * 128 * 4);
  const size_t off_nx2   = take((size_t)BATCH * 128 * 3 * 4);
  const size_t off_l2    = take((size_t)BATCH * 128 * 512 * 4);
  const size_t off_g3    = take((size_t)BATCH * 1024 * 4);
  const size_t off_fc1   = take((size_t)BATCH * 512 * 4);
  const size_t off_fc2   = take((size_t)BATCH * 256 * 4);
  for (int i = 0; i < 21; ++i) {
    int kp = a32h(L[i].cin), cp = a32h(L[i].cout);
    L[i].w_off  = take((size_t)kp * cp * 2);
    L[i].sb_off = take((size_t)2 * cp * 4);
  }
  (void)ws_size;

  // ---- weight / BN prep ----
  for (int i = 0; i < 21; ++i) {
    int kp = a32h(L[i].cin), cp = a32h(L[i].cout);
    int total = kp * cp;
    prep_weight<<<(total + 255) / 256, 256, 0, stream>>>(
        (const float*)d_in[L[i].Wi], L[i].cin, L[i].cout, kp, cp,
        (_Float16*)(ws + L[i].w_off));
    prep_sb<<<(cp + 255) / 256, 256, 0, stream>>>(
        (const float*)d_in[L[i].bi],
        L[i].gi >= 0 ? (const float*)d_in[L[i].gi] : nullptr,
        L[i].bei >= 0 ? (const float*)d_in[L[i].bei] : nullptr,
        L[i].mi >= 0 ? (const float*)d_in[L[i].mi] : nullptr,
        L[i].vi >= 0 ? (const float*)d_in[L[i].vi] : nullptr,
        L[i].cout, cp, (float*)(ws + L[i].sb_off));
  }

  // zero atomic-max targets
  hipMemsetAsync(ws + off_l1, 0, (size_t)BATCH * 512 * 320 * 4, stream);
  hipMemsetAsync(ws + off_l2, 0, (size_t)BATCH * 128 * 512 * 4, stream);
  hipMemsetAsync(ws + off_g3, 0, (size_t)BATCH * 1024 * 4, stream);

  // allow large dynamic LDS for the fused SA kernel (sa3 needs ~205 KB)
  hipFuncSetAttribute(reinterpret_cast<const void*>(sa_group_mlp),
                      hipFuncAttributeMaxDynamicSharedMemorySize, 212992);

  const float* xyz = (const float*)d_in[0];
  float* nx1 = (float*)(ws + off_nx1);
  float* nx2 = (float*)(ws + off_nx2);
  float* l1  = (float*)(ws + off_l1);
  float* l2  = (float*)(ws + off_l2);
  int*   bq  = (int*)(ws + off_bq);

  auto launch_sa = [&](const float* xs, const float* fs, const float* nx,
                       const int* idx, int NP, int Nsrc, int K, int Cfeat,
                       int mode, int l0, float* out, int ostride, int ocoff) {
    const int c0in  = a32h(mode == 0 ? 3 : Cfeat + 3);
    const int c0op  = a32h(L[l0].cout);
    const int c1op  = a32h(L[l0 + 1].cout);
    const int c2op  = a32h(L[l0 + 2].cout);
    const size_t szA = (size_t)K * (c0in > c1op ? c0in : c1op);
    const size_t szB = (size_t)K * c0op;
    const size_t smem = (szA + szB) * sizeof(_Float16);
    sa_group_mlp<<<BATCH * NP, 256, smem, stream>>>(
        xs, fs, nx, idx, NP, Nsrc, K, Cfeat, mode,
        (const _Float16*)(ws + L[l0].w_off),     (const float*)(ws + L[l0].sb_off),
        c0in, L[l0].cout, c0op,
        (const _Float16*)(ws + L[l0 + 1].w_off), (const float*)(ws + L[l0 + 1].sb_off),
        L[l0 + 1].cout, c1op,
        (const _Float16*)(ws + L[l0 + 2].w_off), (const float*)(ws + L[l0 + 2].sb_off),
        L[l0 + 2].cout, c2op,
        (int)szA, out, ostride, ocoff);
  };

  // ---- SA1: FPS(8192 -> 512), 3 scales ----
  fps_kernel<<<BATCH, 256, 0, stream>>>(xyz, 8192, 512,
                                        (float*)(ws + off_dist),
                                        (int*)(ws + off_idxf1));
  gather_xyz<<<(BATCH * 512 + 255) / 256, 256, 0, stream>>>(
      xyz, (const int*)(ws + off_idxf1), 512, 8192, nx1);

  const float r1[3] = {0.1f, 0.2f, 0.4f};
  const int   k1[3] = {32, 64, 128};
  const int   o1[3] = {0, 64, 192};
  for (int s = 0; s < 3; ++s) {
    ball_query_kernel<<<(BATCH * 512 + 255) / 256, 256, 0, stream>>>(
        xyz, nx1, 8192, 512, k1[s], r1[s] * r1[s], bq);
    launch_sa(xyz, nullptr, nx1, bq, 512, 8192, k1[s], 0, 0, s * 3,
              l1, 320, o1[s]);
  }

  // ---- SA2: FPS(512 -> 128), 2 scales ----
  fps_kernel<<<BATCH, 256, 0, stream>>>(nx1, 512, 128,
                                        (float*)(ws + off_dist),
                                        (int*)(ws + off_idxf2));
  gather_xyz<<<(BATCH * 128 + 255) / 256, 256, 0, stream>>>(
      nx1, (const int*)(ws + off_idxf2), 128, 512, nx2);

  const float r2s[2] = {0.4f, 0.8f};
  const int   k2[2] = {64, 128};
  const int   o2[2] = {0, 256};
  for (int s = 0; s < 2; ++s) {
    ball_query_kernel<<<(BATCH * 128 + 255) / 256, 256, 0, stream>>>(
        nx1, nx2, 512, 128, k2[s], r2s[s] * r2s[s], bq);
    launch_sa(nx1, l1, nx2, bq, 128, 512, k2[s], 320, 1, 9 + s * 3,
              l2, 512, o2[s]);
  }

  // ---- SA3: global MLP over 128 points, channels 515 -> 1024, max ----
  launch_sa(nx2, l2, nx2, nullptr, 1, 128, 128, 512, 2, 15,
            (float*)(ws + off_g3), 1024, 0);

  // ---- FC head ----
  dense_layer<<<1, 256, 16 * 1024 * sizeof(_Float16), stream>>>(
      (const float*)(ws + off_g3), BATCH, 1024, 1024,
      (const _Float16*)(ws + L[18].w_off), (const float*)(ws + L[18].sb_off),
      512, 512, (float*)(ws + off_fc1), 1);
  dense_layer<<<1, 256, 16 * 512 * sizeof(_Float16), stream>>>(
      (const float*)(ws + off_fc1), BATCH, 512, 512,
      (const _Float16*)(ws + L[19].w_off), (const float*)(ws + L[19].sb_off),
      256, 256, (float*)(ws + off_fc2), 1);
  dense_layer<<<1, 256, 16 * 256 * sizeof(_Float16), stream>>>(
      (const float*)(ws + off_fc2), BATCH, 256, 256,
      (const _Float16*)(ws + L[20].w_off), (const float*)(ws + L[20].sb_off),
      128, 128, (float*)d_out, 0);
}
